// Network_76493367542191
// MI455X (gfx1250) — matrix operand order repaired and checked by reading, not compiled, not run on hardware
//
// MI455X (gfx1250) implementation: HRR-binding network.
// - E-axis circulant binds + GLU matmul -> v_wmma_f32_16x16x32_bf16 (fp32 accum)
// - A-tiles staged once per block into LDS as bf16 (ds_load_b128 fragment builds)
// - depthwise 128-tap circular seq-conv -> LDS-tiled fp32 VALU
// - LN / GLU / skip fused into GEMM epilogues through LDS.
#include <hip/hip_runtime.h>

#define BB   8
#define SS   16384
#define EE   256
#define LL   2
#define KK   128
#define OO   2
#define ROWS (BB * SS) // 131072
#define APAD 264       // LDS row stride (ushorts) for A tiles: 528B -> bank-conflict free

typedef __attribute__((ext_vector_type(16))) __bf16 v16bf;
typedef __attribute__((ext_vector_type(8)))  float  v8f;

__device__ __forceinline__ unsigned short f2bf_u(float f) {
  unsigned u = __builtin_bit_cast(unsigned, f);
  u += 0x7FFFu + ((u >> 16) & 1u);           // round-to-nearest-even
  return (unsigned short)(u >> 16);
}

// packed pair f32 -> 2x bf16 (uses v_cvt_pk_bf16_f32 when available)
__device__ __forceinline__ unsigned cvt2(float x, float y) {
#if __has_builtin(__builtin_amdgcn_cvt_pk_bf16_f32)
  auto r = __builtin_amdgcn_cvt_pk_bf16_f32(x, y);
  unsigned u;
  __builtin_memcpy(&u, &r, 4);
  return u;
#else
  return (unsigned)f2bf_u(x) | ((unsigned)f2bf_u(y) << 16);
#endif
}

union ABf { unsigned short s[16]; uint4 q[2]; v16bf v; };

// Stage R rows x 256 cols of fp32 A into LDS as bf16 (row stride APAD ushorts).
template <int R>
__device__ __forceinline__ void stage_a(const float* __restrict__ A, int m0blk,
                                        unsigned short (*As)[APAD], int tid) {
  const int pairs = R * 128;
  for (int i = tid; i < pairs; i += 256) {
    int r = i >> 7, cp = i & 127;
    float2 f = reinterpret_cast<const float2*>(A + (size_t)(m0blk + r) * EE)[cp];
    reinterpret_cast<unsigned*>(&As[r][0])[cp] = cvt2(f.x, f.y);
  }
}

// A fragment (16x32 bf16) from LDS bf16 tile.
// lane<16: M=lane, K = kb+{0..7,16..23}; lane>=16: M=lane-16, K = kb+{8..15,24..31}
__device__ __forceinline__ v16bf load_a_lds(const unsigned short (*As)[APAD], int rowLocal,
                                            int kb, int half) {
  const uint4* p = reinterpret_cast<const uint4*>(&As[rowLocal][kb + half * 8]);
  ABf u; u.q[0] = p[0]; u.q[1] = p[2];   // two 16B chunks, 32B apart
  return u.v;
}

// B fragment (32x16 bf16) from N-major bf16 weights BT[n][k], ldb = EE.
// lane<16: N=lane, K = kb+{0..7,16..23}; lane>=16: N=lane-16, K = kb+{8..15,24..31} -> two b128 loads
__device__ __forceinline__ v16bf load_b_bf16(const unsigned short* __restrict__ BT, int n,
                                             int kb, int half) {
  const uint4* p = reinterpret_cast<const uint4*>(BT + (size_t)n * EE + kb + half * 8);
  ABf u; u.q[0] = p[0]; u.q[1] = p[2];
  return u.v;
}

// ---------------- weight preprocessing ----------------
__global__ void k_build_circ(const float* __restrict__ ef, const float* __restrict__ df,
                             unsigned short* __restrict__ MefT, unsigned short* __restrict__ MdfT) {
  int l = blockIdx.x, n = threadIdx.x;
  for (int k = 0; k < EE; ++k) {
    MefT[((size_t)l * EE + n) * EE + k] = f2bf_u(ef[l * EE + ((n - k) & (EE - 1))]);
    MdfT[((size_t)l * EE + n) * EE + k] = f2bf_u(df[l * EE + ((k - n) & (EE - 1))]);
  }
}

__global__ void k_build_w1t(const float* __restrict__ W1, unsigned short* __restrict__ W1T) {
  int idx = blockIdx.x * 256 + threadIdx.x;     // over L*512*256
  int l = idx / (512 * EE);
  int rem = idx - l * 512 * EE;
  int n = rem >> 8, k = rem & 255;
  W1T[idx] = f2bf_u(W1[(size_t)l * EE * 512 + (size_t)k * 512 + n]);
}

__global__ void k_build_kn(const float* __restrict__ cf, float* __restrict__ kn) {
  int l = blockIdx.x, e = threadIdx.x;
  float s = 0.f;
  for (int t = 0; t < KK; ++t) { float v = cf[((size_t)l * KK + t) * EE + e]; s += v * v; }
  float inv = 1.0f / sqrtf(s);
  for (int t = 0; t < KK; ++t)
    kn[((size_t)l * KK + t) * EE + e] = cf[((size_t)l * KK + t) * EE + e] * inv;
}

// ---------------- stage 0: embed + pos + LN ----------------
__global__ __launch_bounds__(256) void k_embed_ln(const int* __restrict__ tok,
                                                  const float* __restrict__ emb,
                                                  const float* __restrict__ pos,
                                                  const float* __restrict__ gs,
                                                  const float* __restrict__ gb,
                                                  float* __restrict__ X) {
  int tid = threadIdx.x, wid = tid >> 5, lid = tid & 31;
  int row = blockIdx.x * 8 + wid;
  int s = row & (SS - 1);
  int t = tok[row];
  float v[8], sum = 0.f;
#pragma unroll
  for (int j = 0; j < 8; ++j) {
    int e = lid + j * 32;
    v[j] = emb[(size_t)t * EE + e] + pos[(size_t)s * EE + e];
    sum += v[j];
  }
  for (int off = 16; off > 0; off >>= 1) sum += __shfl_xor(sum, off, 32);
  float mean = sum * (1.0f / EE);
  float sq = 0.f;
#pragma unroll
  for (int j = 0; j < 8; ++j) { float d = v[j] - mean; sq += d * d; }
  for (int off = 16; off > 0; off >>= 1) sq += __shfl_xor(sq, off, 32);
  float rstd = rsqrtf(sq * (1.0f / EE) + 1e-6f);
#pragma unroll
  for (int j = 0; j < 8; ++j) {
    int e = lid + j * 32;
    X[(size_t)row * EE + e] = (v[j] - mean) * rstd * gs[e] + gb[e];
  }
}

// ---------------- stage 1: bind along E  (GEMM 131072x256x256) ----------------
__global__ __launch_bounds__(256) void k_gemm_bind(const float* __restrict__ A,
                                                   const unsigned short* __restrict__ BT,
                                                   float* __restrict__ C) {
  __shared__ __align__(16) unsigned short As[32][APAD];
  int tid = threadIdx.x, wave = tid >> 5, lid = tid & 31;
  int half = lid >> 4, lane16 = lid & 15;
  int wm = wave >> 2, wn = wave & 3;
  int mb = blockIdx.x * 32;
  int m0 = mb + wm * 16;
  int n0 = wn * 64;
  stage_a<32>(A, mb, As, tid);
  __syncthreads();
  v8f acc[4] = {};
  for (int kb = 0; kb < EE; kb += 32) {
    v16bf a = load_a_lds(As, wm * 16 + lane16, kb, half);
    v16bf b0 = load_b_bf16(BT, n0 + 0  + lane16, kb, half);
    v16bf b1 = load_b_bf16(BT, n0 + 16 + lane16, kb, half);
    v16bf b2 = load_b_bf16(BT, n0 + 32 + lane16, kb, half);
    v16bf b3 = load_b_bf16(BT, n0 + 48 + lane16, kb, half);
    acc[0] = __builtin_amdgcn_wmma_f32_16x16x32_bf16(false, a, false, b0, (short)0, acc[0], false, false);
    acc[1] = __builtin_amdgcn_wmma_f32_16x16x32_bf16(false, a, false, b1, (short)0, acc[1], false, false);
    acc[2] = __builtin_amdgcn_wmma_f32_16x16x32_bf16(false, a, false, b2, (short)0, acc[2], false, false);
    acc[3] = __builtin_amdgcn_wmma_f32_16x16x32_bf16(false, a, false, b3, (short)0, acc[3], false, false);
  }
  int r0 = m0 + half * 8;
#pragma unroll
  for (int t = 0; t < 4; ++t) {
    int col = n0 + t * 16 + lane16;
#pragma unroll
    for (int j = 0; j < 8; ++j) C[(size_t)(r0 + j) * EE + col] = acc[t][j];
  }
}

// ------- stage 2: depthwise 128-tap circular conv along S + x*w + gelu(tanh) -------
__global__ __launch_bounds__(256) void k_conv_gelu(const float* __restrict__ in,
                                                   const float* __restrict__ knl,
                                                   const float* __restrict__ wl,
                                                   float* __restrict__ outp) {
  __shared__ float tile[192][32];   // window [s0-127, s0+63] x 32 channels
  __shared__ float kt[KK][32];
  int tid = threadIdx.x;
  int bi = blockIdx.x;              // B * S/64 blocks
  int b = bi >> 8;
  int s0 = (bi & 255) * 64;
  int e0 = blockIdx.y * 32;
  for (int i = tid; i < 191 * 32; i += 256) {
    int sr = i >> 5, ch = i & 31;
    int s = (s0 - 127 + sr) & (SS - 1);       // circular wrap
    tile[sr][ch] = in[((size_t)b * SS + s) * EE + e0 + ch];
  }
  for (int i = tid; i < KK * 32; i += 256) {
    int t = i >> 5, ch = i & 31;
    kt[t][ch] = knl[(size_t)t * EE + e0 + ch];
  }
  __syncthreads();
  int ch = tid & 31, sg = tid >> 5;
  float wv = wl[e0 + ch];
  float acc[8];
#pragma unroll
  for (int r = 0; r < 8; ++r) acc[r] = 0.f;
  int base = sg * 8 + 127;
  for (int t = 0; t < KK; ++t) {
    float kv = kt[t][ch];
    int bb = base - t;
#pragma unroll
    for (int r = 0; r < 8; ++r) acc[r] += tile[bb + r][ch] * kv;
  }
#pragma unroll
  for (int r = 0; r < 8; ++r) {
    float xv = tile[base + r][ch];
    float y = acc[r] + xv * wv;
    float u = 0.7978845608028654f * (y + 0.044715f * y * y * y);
    float g = 0.5f * y * (1.0f + tanhf(u));
    outp[((size_t)b * SS + s0 + sg * 8 + r) * EE + e0 + ch] = g;
  }
}

// ------- stage 3: unbind along E (GEMM) + LayerNorm fused, in-place -------
__global__ __launch_bounds__(256) void k_unbind_ln(float* __restrict__ X, // in & out
                                                   const unsigned short* __restrict__ BT,
                                                   const float* __restrict__ gs,
                                                   const float* __restrict__ gb) {
  __shared__ __align__(16) unsigned short As[32][APAD];
  __shared__ float tile[32][EE + 4];
  __shared__ float r1[32][8], r2[32][8];
  __shared__ float mrow[32], rrow[32];
  int tid = threadIdx.x, wave = tid >> 5, lid = tid & 31;
  int half = lid >> 4, lane16 = lid & 15;
  int wm = wave >> 2, wn = wave & 3;
  int mb = blockIdx.x * 32;
  int n0 = wn * 64;
  stage_a<32>(X, mb, As, tid);
  __syncthreads();
  v8f acc[4] = {};
  for (int kb = 0; kb < EE; kb += 32) {
    v16bf a = load_a_lds(As, wm * 16 + lane16, kb, half);
    v16bf b0 = load_b_bf16(BT, n0 + 0  + lane16, kb, half);
    v16bf b1 = load_b_bf16(BT, n0 + 16 + lane16, kb, half);
    v16bf b2 = load_b_bf16(BT, n0 + 32 + lane16, kb, half);
    v16bf b3 = load_b_bf16(BT, n0 + 48 + lane16, kb, half);
    acc[0] = __builtin_amdgcn_wmma_f32_16x16x32_bf16(false, a, false, b0, (short)0, acc[0], false, false);
    acc[1] = __builtin_amdgcn_wmma_f32_16x16x32_bf16(false, a, false, b1, (short)0, acc[1], false, false);
    acc[2] = __builtin_amdgcn_wmma_f32_16x16x32_bf16(false, a, false, b2, (short)0, acc[2], false, false);
    acc[3] = __builtin_amdgcn_wmma_f32_16x16x32_bf16(false, a, false, b3, (short)0, acc[3], false, false);
  }
  int rl = wm * 16 + half * 8;
#pragma unroll
  for (int t = 0; t < 4; ++t) {
    int col = n0 + t * 16 + lane16;
#pragma unroll
    for (int j = 0; j < 8; ++j) tile[rl + j][col] = acc[t][j];
  }
  __syncthreads();
  int row = tid >> 3, prt = tid & 7;
  float s1 = 0.f, s2 = 0.f;
  for (int i = 0; i < 32; ++i) { float v = tile[row][prt * 32 + i]; s1 += v; s2 += v * v; }
  r1[row][prt] = s1; r2[row][prt] = s2;
  __syncthreads();
  if (prt == 0) {
    float a1 = 0.f, a2 = 0.f;
    for (int i = 0; i < 8; ++i) { a1 += r1[row][i]; a2 += r2[row][i]; }
    float mean = a1 * (1.0f / EE);
    float var = a2 * (1.0f / EE) - mean * mean;
    mrow[row] = mean; rrow[row] = rsqrtf(var + 1e-6f);
  }
  __syncthreads();
  float sg = gs[tid], bg = gb[tid];             // col == tid
  for (int i = 0; i < 32; ++i) {
    float v = tile[i][tid];
    X[(size_t)(mb + i) * EE + tid] = (v - mrow[i]) * rrow[i] * sg + bg;
  }
}

// ------- stage 4: h = x @ W1 + b1 ; GLU ; + skip (writes skip buffer in place) -------
__global__ __launch_bounds__(256) void k_glu_skip(const float* __restrict__ A,
                                                  const unsigned short* __restrict__ BT,
                                                  const float* __restrict__ bias,
                                                  float* __restrict__ Xio) {
  __shared__ __align__(16) unsigned short As[16][APAD];
  __shared__ float h[16][512 + 4];
  int tid = threadIdx.x, wave = tid >> 5, lid = tid & 31;
  int half = lid >> 4, lane16 = lid & 15;
  int m0 = blockIdx.x * 16;
  int n0 = wave * 64;                           // 8 waves cover N=512
  stage_a<16>(A, m0, As, tid);
  __syncthreads();
  v8f acc[4] = {};
  for (int kb = 0; kb < EE; kb += 32) {
    v16bf a = load_a_lds(As, lane16, kb, half);
    v16bf b0 = load_b_bf16(BT, n0 + 0  + lane16, kb, half);
    v16bf b1 = load_b_bf16(BT, n0 + 16 + lane16, kb, half);
    v16bf b2 = load_b_bf16(BT, n0 + 32 + lane16, kb, half);
    v16bf b3 = load_b_bf16(BT, n0 + 48 + lane16, kb, half);
    acc[0] = __builtin_amdgcn_wmma_f32_16x16x32_bf16(false, a, false, b0, (short)0, acc[0], false, false);
    acc[1] = __builtin_amdgcn_wmma_f32_16x16x32_bf16(false, a, false, b1, (short)0, acc[1], false, false);
    acc[2] = __builtin_amdgcn_wmma_f32_16x16x32_bf16(false, a, false, b2, (short)0, acc[2], false, false);
    acc[3] = __builtin_amdgcn_wmma_f32_16x16x32_bf16(false, a, false, b3, (short)0, acc[3], false, false);
  }
  int rl = half * 8;
#pragma unroll
  for (int t = 0; t < 4; ++t) {
    int col = n0 + t * 16 + lane16;
    float bv = bias[col];
#pragma unroll
    for (int j = 0; j < 8; ++j) h[rl + j][col] = acc[t][j] + bv;
  }
  __syncthreads();
  for (int i = 0; i < 16; ++i) {
    float av = h[i][tid];
    float gv = h[i][tid + 256];
    float sig = 1.0f / (1.0f + expf(-gv));
    size_t off = (size_t)(m0 + i) * EE + tid;
    Xio[off] = Xio[off] + av * sig;
  }
}

// ---------------- head: mean over S, logits, log_softmax ----------------
__global__ __launch_bounds__(256) void k_meanpart(const float* __restrict__ X,
                                                  float* __restrict__ part) {
  int b = blockIdx.x, c = blockIdx.y, e = threadIdx.x;
  float s = 0.f;
  const float* p = X + ((size_t)b * SS + (size_t)c * 256) * EE + e;
  for (int i = 0; i < 256; ++i) s += p[(size_t)i * EE];
  part[((size_t)(b * 64 + c)) * EE + e] = s;
}

__global__ __launch_bounds__(256) void k_head(const float* __restrict__ part,
                                              const float* __restrict__ Wout,
                                              const float* __restrict__ bout,
                                              float* __restrict__ out) {
  __shared__ float red0[256];
  __shared__ float red1[256];
  int tid = threadIdx.x;
  for (int b = 0; b < BB; ++b) {
    float m = 0.f;
    for (int c = 0; c < 64; ++c) m += part[((size_t)(b * 64 + c)) * EE + tid];
    m *= (1.0f / (float)SS);
    red0[tid] = m * Wout[tid * OO + 0];
    red1[tid] = m * Wout[tid * OO + 1];
    __syncthreads();
    for (int off = 128; off > 0; off >>= 1) {
      if (tid < off) { red0[tid] += red0[tid + off]; red1[tid] += red1[tid + off]; }
      __syncthreads();
    }
    if (tid == 0) {
      float l0 = red0[0] + bout[0], l1 = red1[0] + bout[1];
      float mx = fmaxf(l0, l1);
      float lse = mx + logf(expf(l0 - mx) + expf(l1 - mx));
      out[b * OO + 0] = l0 - lse;
      out[b * OO + 1] = l1 - lse;
    }
    __syncthreads();
  }
}

extern "C" void kernel_launch(void* const* d_in, const int* in_sizes, int n_in,
                              void* d_out, int out_size, void* d_ws, size_t ws_size,
                              hipStream_t stream) {
  const int*   en_in = (const int*)d_in[0];
  const float* emb   = (const float*)d_in[1];
  const float* pos   = (const float*)d_in[2];
  const float* ln0s  = (const float*)d_in[3];
  const float* ln0b  = (const float*)d_in[4];
  const float* ef    = (const float*)d_in[5];
  const float* cf    = (const float*)d_in[6];
  const float* df    = (const float*)d_in[7];
  const float* w     = (const float*)d_in[8];
  const float* lns   = (const float*)d_in[9];
  const float* lnb   = (const float*)d_in[10];
  const float* W1    = (const float*)d_in[11];
  const float* b1    = (const float*)d_in[12];
  const float* Wout  = (const float*)d_in[13];
  const float* bout  = (const float*)d_in[14];
  float* out = (float*)d_out;

  char* ws = (char*)d_ws;
  const size_t XB = (size_t)ROWS * EE * sizeof(float);        // 134 MB
  float* bufX  = (float*)(ws);                                 // x / skip (in-place updated)
  float* bufT  = (float*)(ws + XB);                            // y1
  float* bufT2 = (float*)(ws + 2 * XB);                        // gelu out / x3
  char* p = ws + 3 * XB;
  unsigned short* MefT = (unsigned short*)p; p += (size_t)LL * EE * EE * 2;
  unsigned short* MdfT = (unsigned short*)p; p += (size_t)LL * EE * EE * 2;
  unsigned short* W1T  = (unsigned short*)p; p += (size_t)LL * 2 * EE * EE * 2;
  float* kn      = (float*)p; p += (size_t)LL * KK * EE * 4;
  float* partial = (float*)p; p += (size_t)BB * 64 * EE * 4;

  k_build_circ<<<LL, 256, 0, stream>>>(ef, df, MefT, MdfT);
  k_build_w1t<<<(LL * 2 * EE * EE) / 256, 256, 0, stream>>>(W1, W1T);
  k_build_kn<<<LL, 256, 0, stream>>>(cf, kn);
  k_embed_ln<<<ROWS / 8, 256, 0, stream>>>(en_in, emb, pos, ln0s, ln0b, bufX);

  for (int l = 0; l < LL; ++l) {
    k_gemm_bind<<<ROWS / 32, 256, 0, stream>>>(bufX, MefT + (size_t)l * EE * EE, bufT);
    k_conv_gelu<<<dim3((BB * SS) / 64, EE / 32), 256, 0, stream>>>(
        bufT, kn + (size_t)l * KK * EE, w + l * EE, bufT2);
    k_unbind_ln<<<ROWS / 32, 256, 0, stream>>>(bufT2, MdfT + (size_t)l * EE * EE,
                                               lns + l * EE, lnb + l * EE);
    k_glu_skip<<<ROWS / 16, 256, 0, stream>>>(bufT2, W1T + (size_t)l * 2 * EE * EE,
                                              b1 + l * 2 * EE, bufX);
  }

  k_meanpart<<<dim3(BB, 64), 256, 0, stream>>>(bufX, partial);
  k_head<<<1, 256, 0, stream>>>(partial, Wout, bout, out);
}
